// Attention_14577119003244
// MI455X (gfx1250) — compile-verified, matrix operand order checked by reading, not run on hardware
//
#include <hip/hip_runtime.h>
#include <hip/hip_bf16.h>

// ---------------------------------------------------------------------------
// Causal MHA forward for MI455X (gfx1250, wave32, WMMA).
// All matmuls run on v_wmma_f32_16x16x32_bf16 (bf16 in, fp32 accumulate).
// GEMM staging uses GLOBAL_LOAD_ASYNC_TO_LDS_B128 (ASYNCcnt) with LDS
// double-buffering when the toolchain exposes the builtin.
// ---------------------------------------------------------------------------

using bf16 = __bf16;
typedef __attribute__((ext_vector_type(16))) __bf16 v16bf;
typedef __attribute__((ext_vector_type(8)))  __bf16 v8bf;
typedef __attribute__((ext_vector_type(8)))  float  v8f;
typedef int v4i_vs __attribute__((vector_size(16)));  // matches builtin param

#define LOG_BASE 9.210340371976184f  // ln(10000)

#if defined(__has_builtin)
#if __has_builtin(__builtin_amdgcn_global_load_async_to_lds_b128)
#define HAVE_ASYNC_CP 1
#endif
#if __has_builtin(__builtin_amdgcn_s_wait_asynccnt)
#define HAVE_WAIT_ASYNC 1
#endif
#endif
#ifndef HAVE_ASYNC_CP
#define HAVE_ASYNC_CP 0
#endif
#ifndef HAVE_WAIT_ASYNC
#define HAVE_WAIT_ASYNC 0
#endif

#define GLOBAL_AS __attribute__((address_space(1)))
#define LDS_AS    __attribute__((address_space(3)))

// 16-byte global -> LDS copy. Async (ASYNCcnt, no VGPR round-trip) when
// available, otherwise synchronous load+ds_store.
__device__ __forceinline__ void cp16_g2l(const bf16* g, bf16* l) {
#if HAVE_ASYNC_CP
  __builtin_amdgcn_global_load_async_to_lds_b128(
      (GLOBAL_AS v4i_vs*)g, (LDS_AS v4i_vs*)l, 0, 0);
#else
  *(v8bf*)l = *(const v8bf*)g;
#endif
}

__device__ __forceinline__ void wait_async_cp() {
#if HAVE_ASYNC_CP
#if HAVE_WAIT_ASYNC
  __builtin_amdgcn_s_wait_asynccnt(0);
#else
  asm volatile("s_wait_asynccnt 0x0" ::: "memory");
#endif
#endif
}

// Build a 16x32 A/B fragment (wave32 layout) from two 16-byte chunks:
// elements 0..7 = K kb..kb+7, elements 8..15 = K kb+16..kb+23.
__device__ __forceinline__ v16bf frag_ld(const bf16* lo, const bf16* hi) {
  v8bf a = *(const v8bf*)lo;
  v8bf b = *(const v8bf*)hi;
  return __builtin_shufflevector(a, b, 0,1,2,3,4,5,6,7,8,9,10,11,12,13,14,15);
}

__device__ __forceinline__ v8f wmma_bf16(v16bf a, v16bf b, v8f c) {
  // (neg_a, A, neg_b, B, c_mod, C, reuse_a, reuse_b)
  return __builtin_amdgcn_wmma_f32_16x16x32_bf16(false, a, false, b, (short)0, c,
                                                 false, false);
}

// ---------------------------------------------------------------------------
// fp32 -> bf16 elementwise convert (grid-stride)
// ---------------------------------------------------------------------------
__global__ __launch_bounds__(256)
void f2b_kernel(const float* __restrict__ in, bf16* __restrict__ out, int n) {
  for (int i = blockIdx.x * 256 + threadIdx.x; i < n; i += gridDim.x * 256)
    out[i] = (bf16)in[i];
}

// ---------------------------------------------------------------------------
// C[M,N] (fp32) = A[M,K](bf16) * B[N,K](bf16)^T  -- both operands k-contiguous.
// 128x128 tile per 256-thread block; 8 waves, each owning a 16x128 strip.
// Double-buffered LDS, async global->LDS staging overlapped with WMMA.
// ---------------------------------------------------------------------------
__global__ __launch_bounds__(256)
void gemm_bf16_nt(const bf16* __restrict__ A, const bf16* __restrict__ B,
                  float* __restrict__ C, int M, int N, int K) {
  constexpr int LDT = 40;  // 32 + 8 bf16 pad (80B row stride, 16B aligned)
  __shared__ __align__(16) bf16 As[2][128 * LDT];
  __shared__ __align__(16) bf16 Bs[2][128 * LDT];

  const int mBase = blockIdx.y * 128;
  const int nBase = blockIdx.x * 128;
  const int w    = threadIdx.x >> 5;
  const int lane = threadIdx.x & 31;
  const int lr   = lane & 15;
  const int hi   = lane >> 4;
  const int kb   = hi * 8;  // fragment K sub-base per half-wave

  // Stage one 128x32 bf16 tile pair into LDS buffer `buf` (16B chunks).
  auto stage = [&](int buf, int kk) {
#pragma unroll
    for (int it = 0; it < 2; ++it) {
      const int c    = threadIdx.x + it * 256;  // 0..511
      const int row  = c >> 2;
      const int col8 = (c & 3) * 8;
      cp16_g2l(A + (size_t)(mBase + row) * K + kk + col8,
               &As[buf][row * LDT + col8]);
      cp16_g2l(B + (size_t)(nBase + row) * K + kk + col8,
               &Bs[buf][row * LDT + col8]);
    }
  };

  v8f acc[8] = {};
  const int nTiles = K / 32;
  stage(0, 0);

  for (int t = 0; t < nTiles; ++t) {
    wait_async_cp();
    __syncthreads();  // tile t resident in buf t&1; all waves past tile t-1
    if (t + 1 < nTiles) stage((t + 1) & 1, (t + 1) * 32);  // overlaps compute

    const bf16* as = As[t & 1];
    const bf16* bs = Bs[t & 1];
    const int ar = (w * 16 + lr) * LDT + kb;
    v16bf a = frag_ld(&as[ar], &as[ar + 16]);
    v16bf bfr[8];
#pragma unroll
    for (int j = 0; j < 8; ++j) {
      const int br = (j * 16 + lr) * LDT + kb;
      bfr[j] = frag_ld(&bs[br], &bs[br + 16]);
    }
#pragma unroll
    for (int j = 0; j < 8; ++j) acc[j] = wmma_bf16(a, bfr[j], acc[j]);
  }

#pragma unroll
  for (int j = 0; j < 8; ++j)
#pragma unroll
    for (int r = 0; r < 8; ++r)
      C[(size_t)(mBase + w * 16 + r + 8 * hi) * N + nBase + j * 16 + lr] =
          acc[j][r];
}

// ---------------------------------------------------------------------------
// RoPE + repack: fp32 [B*S, D] -> bf16 [B, H, S, 64].
// One thread per (b, s, h, pair i in 0..31).
// ---------------------------------------------------------------------------
__global__ __launch_bounds__(256)
void rope_pack(const float* __restrict__ Qf, const float* __restrict__ Kf,
               const float* __restrict__ Vf, bf16* __restrict__ Qb,
               bf16* __restrict__ Kb, bf16* __restrict__ Vb) {
  const int tid = blockIdx.x * 256 + threadIdx.x;
  const int i = tid & 31;
  const int h = (tid >> 5) & 15;
  const int s = (tid >> 9) & 2047;
  const int b = tid >> 20;

  const size_t in0  = ((size_t)(b * 2048 + s)) * 1024 + h * 64 + 2 * i;
  const size_t out0 = (((size_t)(b * 16 + h)) * 2048 + s) * 64 + 2 * i;

  const float pos = (float)s;
  const int j0 = 2 * i, j1 = 2 * i + 1;
  const float f0 = __expf(-((float)(2 * (j0 & 31)) * (1.0f / 64.0f)) * LOG_BASE);
  const float f1 = __expf(-((float)(2 * (j1 & 31)) * (1.0f / 64.0f)) * LOG_BASE);
  const float c0 = __cosf(pos * f0), sn0 = __sinf(pos * f0);
  const float c1 = __cosf(pos * f1), sn1 = __sinf(pos * f1);

  const float q0 = Qf[in0], q1 = Qf[in0 + 1];
  const float k0 = Kf[in0], k1 = Kf[in0 + 1];
  // x_rot[2i] = -x[2i+1], x_rot[2i+1] = x[2i]
  Qb[out0]     = (bf16)(q0 * c0 - q1 * sn0);
  Qb[out0 + 1] = (bf16)(q1 * c1 + q0 * sn1);
  Kb[out0]     = (bf16)(k0 * c0 - k1 * sn0);
  Kb[out0 + 1] = (bf16)(k1 * c1 + k0 * sn1);
  Vb[out0]     = (bf16)Vf[in0];
  Vb[out0 + 1] = (bf16)Vf[in0 + 1];
}

// ---------------------------------------------------------------------------
// Flash attention with online softmax.
// grid = (S/128, B*H), block = 256 (8 waves). Wave w owns 16 query rows.
// Key blocks of 32; scores: 2 tiles x 2 WMMA; PV: 4 WMMA.
// K tile staged via async copy; V transposed through VGPRs.
// Out written fp32 as [B, S, H*64] (token-major) for the final GEMM.
// ---------------------------------------------------------------------------
__global__ __launch_bounds__(256)
void attn_flash(const bf16* __restrict__ Q, const bf16* __restrict__ K,
                const bf16* __restrict__ V, float* __restrict__ Out) {
  constexpr int S = 2048, HD = 64, D = 1024;
  __shared__ __align__(16) bf16 Ks[32 * 72];      // [key][d], +8 pad
  __shared__ __align__(16) bf16 Vs[64 * 40];      // V^T: [d][key], +8 pad
  __shared__ __align__(16) bf16 Ps[8 * 16 * 40];  // per-wave P scratch

  const int bh   = blockIdx.y;
  const int qblk = blockIdx.x;
  const int w    = threadIdx.x >> 5;
  const int lane = threadIdx.x & 31;
  const int lr   = lane & 15;
  const int hi   = lane >> 4;
  const int kb   = hi * 8;
  const int qbase = qblk * 128 + w * 16;

  // Q fragments (16 rows x 64 d) held in registers: 2 x v16bf.
  const bf16* Qp = Q + (size_t)bh * S * HD;
  const int qrowA = qbase + lr;
  v16bf qa[2];
#pragma unroll
  for (int c = 0; c < 2; ++c)
    qa[c] = frag_ld(Qp + (size_t)qrowA * HD + c * 32 + kb,
                    Qp + (size_t)qrowA * HD + c * 32 + kb + 16);

  float Mrow[8], Lrow[8];
  v8f o[4] = {};
#pragma unroll
  for (int r = 0; r < 8; ++r) { Mrow[r] = -1e30f; Lrow[r] = 0.0f; }

  const int nblk = (qblk + 1) * 4;  // key blocks up to the diagonal
  for (int t = 0; t < nblk; ++t) {
    const int k0 = t * 32;
    __syncthreads();  // previous block fully consumed
    {  // stage K (async, row-major) and V (transposed via VGPRs)
      const int c    = threadIdx.x;  // 256 chunks of 8 elems = 32x64
      const int row  = c >> 3;
      const int col8 = (c & 7) * 8;
      cp16_g2l(K + ((size_t)bh * S + k0 + row) * HD + col8,
               &Ks[row * 72 + col8]);
      v8bf vv = *(const v8bf*)(V + ((size_t)bh * S + k0 + row) * HD + col8);
#pragma unroll
      for (int e = 0; e < 8; ++e) Vs[(col8 + e) * 40 + row] = vv[e];
    }
    wait_async_cp();
    __syncthreads();

    // Scores: S[16x32] = Q(16x64) . K^T   (batch frag loads, then WMMA chain)
    v16bf kf[4];
    kf[0] = frag_ld(&Ks[lr * 72 + kb],             &Ks[lr * 72 + kb + 16]);
    kf[1] = frag_ld(&Ks[lr * 72 + 32 + kb],        &Ks[lr * 72 + 32 + kb + 16]);
    kf[2] = frag_ld(&Ks[(16 + lr) * 72 + kb],      &Ks[(16 + lr) * 72 + kb + 16]);
    kf[3] = frag_ld(&Ks[(16 + lr) * 72 + 32 + kb], &Ks[(16 + lr) * 72 + 32 + kb + 16]);
    v8f s0 = {}, s1 = {};
    s0 = wmma_bf16(qa[0], kf[0], s0);
    s1 = wmma_bf16(qa[0], kf[2], s1);
    s0 = wmma_bf16(qa[1], kf[1], s0);
    s1 = wmma_bf16(qa[1], kf[3], s1);

    // Scale + causal mask
#pragma unroll
    for (int r = 0; r < 8; ++r) {
      const int qi = qbase + r + 8 * hi;
      float v0 = s0[r] * 0.125f;  // 1/sqrt(64)
      float v1 = s1[r] * 0.125f;
      if (k0 + lr > qi) v0 = -1e30f;
      if (k0 + 16 + lr > qi) v1 = -1e30f;
      s0[r] = v0;
      s1[r] = v1;
    }

    // Online softmax (row = VGPR index within each 16-lane half)
    float alpha[8];
#pragma unroll
    for (int r = 0; r < 8; ++r) {
      float m = fmaxf(s0[r], s1[r]);
#pragma unroll
      for (int d = 1; d < 16; d <<= 1) m = fmaxf(m, __shfl_xor(m, d, 32));
      const float mn = fmaxf(Mrow[r], m);
      alpha[r] = __expf(Mrow[r] - mn);
      const float p0 = __expf(s0[r] - mn);
      const float p1 = __expf(s1[r] - mn);
      s0[r] = p0;
      s1[r] = p1;
      float l = p0 + p1;
#pragma unroll
      for (int d = 1; d < 16; d <<= 1) l += __shfl_xor(l, d, 32);
      Lrow[r] = Lrow[r] * alpha[r] + l;
      Mrow[r] = mn;
    }
#pragma unroll
    for (int g = 0; g < 4; ++g)
#pragma unroll
      for (int r = 0; r < 8; ++r) o[g][r] *= alpha[r];

    // C-layout fp32 -> A-layout bf16 via wave-private LDS round trip
    bf16* Pw = &Ps[w * 16 * 40];
#pragma unroll
    for (int r = 0; r < 8; ++r) {
      const int prow = r + 8 * hi;
      Pw[prow * 40 + lr]      = (bf16)s0[r];
      Pw[prow * 40 + 16 + lr] = (bf16)s1[r];
    }
    v16bf pa = frag_ld(&Pw[lr * 40 + kb], &Pw[lr * 40 + kb + 16]);

    // O(16x64) += P(16x32) . V(32x64)   (V^T staged -> k-contiguous B frags)
    v16bf vf[4];
#pragma unroll
    for (int g = 0; g < 4; ++g)
      vf[g] = frag_ld(&Vs[(g * 16 + lr) * 40 + kb],
                      &Vs[(g * 16 + lr) * 40 + kb + 16]);
#pragma unroll
    for (int g = 0; g < 4; ++g) o[g] = wmma_bf16(pa, vf[g], o[g]);
  }

  // Normalize and write [B, S, H*64] fp32
  const int b = bh >> 4, h = bh & 15;
#pragma unroll
  for (int g = 0; g < 4; ++g)
#pragma unroll
    for (int r = 0; r < 8; ++r) {
      const int qi = qbase + r + 8 * hi;
      Out[((size_t)b * S + qi) * D + h * 64 + g * 16 + lr] = o[g][r] / Lrow[r];
    }
}

// ---------------------------------------------------------------------------
// Host-side orchestration
// ---------------------------------------------------------------------------
extern "C" void kernel_launch(void* const* d_in, const int* in_sizes, int n_in,
                              void* d_out, int out_size, void* d_ws,
                              size_t ws_size, hipStream_t stream) {
  (void)in_sizes; (void)n_in; (void)out_size; (void)ws_size;
  const float* x  = (const float*)d_in[0];
  const float* Wq = (const float*)d_in[1];
  const float* Wk = (const float*)d_in[2];
  const float* Wv = (const float*)d_in[3];
  const float* Wo = (const float*)d_in[4];
  // d_in[5] (mask) unused: causality computed analytically.
  float* out = (float*)d_out;

  constexpr size_t T = 4ull * 2048;  // tokens
  constexpr size_t D = 1024;

  char* p = (char*)d_ws;
  auto take = [&](size_t bytes) -> char* {
    char* r = p;
    p += (bytes + 255) & ~(size_t)255;
    return r;
  };
  bf16*  xb  = (bf16*)take(T * D * 2);
  bf16*  wqb = (bf16*)take(D * D * 2);
  bf16*  wkb = (bf16*)take(D * D * 2);
  bf16*  wvb = (bf16*)take(D * D * 2);
  bf16*  wob = (bf16*)take(D * D * 2);
  float* Qf  = (float*)take(T * D * 4);
  float* Kf  = (float*)take(T * D * 4);
  float* Vf  = (float*)take(T * D * 4);
  bf16*  Qb  = (bf16*)take(T * D * 2);
  bf16*  Kb  = (bf16*)take(T * D * 2);
  bf16*  Vb  = (bf16*)take(T * D * 2);
  float* Af  = (float*)take(T * D * 4);
  bf16*  Ab  = (bf16*)take(T * D * 2);

  const int nx = (int)(T * D);  // 8M
  const int nw = (int)(D * D);  // 1M
  f2b_kernel<<<4096, 256, 0, stream>>>(x, xb, nx);
  f2b_kernel<<<2048, 256, 0, stream>>>(Wq, wqb, nw);
  f2b_kernel<<<2048, 256, 0, stream>>>(Wk, wkb, nw);
  f2b_kernel<<<2048, 256, 0, stream>>>(Wv, wvb, nw);
  f2b_kernel<<<2048, 256, 0, stream>>>(Wo, wob, nw);

  dim3 gGemm(D / 128, T / 128);  // (8, 64)
  gemm_bf16_nt<<<gGemm, 256, 0, stream>>>(xb, wqb, Qf, (int)T, (int)D, (int)D);
  gemm_bf16_nt<<<gGemm, 256, 0, stream>>>(xb, wkb, Kf, (int)T, (int)D, (int)D);
  gemm_bf16_nt<<<gGemm, 256, 0, stream>>>(xb, wvb, Vf, (int)T, (int)D, (int)D);

  rope_pack<<<16384, 256, 0, stream>>>(Qf, Kf, Vf, Qb, Kb, Vb);

  dim3 gAttn(2048 / 128, 4 * 16);  // (16, 64)
  attn_flash<<<gAttn, 256, 0, stream>>>(Qb, Kb, Vb, Af);

  f2b_kernel<<<4096, 256, 0, stream>>>(Af, Ab, nx);
  gemm_bf16_nt<<<gGemm, 256, 0, stream>>>(Ab, wob, out, (int)T, (int)D, (int)D);
}
